// DiffFlexSelfAttention_10222022164657
// MI455X (gfx1250) — matrix operand order repaired and checked by reading, not compiled, not run on hardware
//
#include <hip/hip_runtime.h>
#include <math.h>

// ---------------- CDNA5 WMMA types ----------------
typedef __attribute__((ext_vector_type(16))) __bf16 v16bf;
typedef __attribute__((ext_vector_type(8)))  __bf16 bf16x8;
typedef __attribute__((ext_vector_type(8)))  float  v8f;

#define S_LEN        2048
#define HID          2048
#define NHEADS       16      // heads per attention branch
#define HD           128     // value head dim
#define QD           64      // q/k head dim
#define SCALE        0.125f  // 1/sqrt(QD)
#define LAMBDA_INIT  0.78360578f
#define F32_EPS      1.1920929e-7f

// WMMA bf16 helper: D = A*B + C  (16x16x32, f32 accum)
__device__ __forceinline__ v8f wmma_bf16(v16bf a, v16bf b, v8f c) {
    return __builtin_amdgcn_wmma_f32_16x16x32_bf16(false, a, false, b,
                                                   (short)0, c, false, false);
}

// A-fragment: 16x32 bf16, row per lane%16; lane<16 holds K[0..7],K[16..23],
// lane>=16 holds K[8..15],K[24..31] (ISA 7.12.2). Row-major source, ld in elems.
__device__ __forceinline__ v16bf load_frag_a(const __bf16* __restrict__ A,
                                             int lda, int m0, int k0) {
    int lane = threadIdx.x & 31;
    int half = lane >> 4, r = lane & 15;
    const __bf16* p = A + (size_t)(m0 + r) * lda + k0 + half * 8;
    bf16x8 c0 = *(const bf16x8*)(p);
    bf16x8 c1 = *(const bf16x8*)(p + 16);
    v16bf v;
#pragma unroll
    for (int i = 0; i < 8; ++i) { v[i] = c0[i]; v[i + 8] = c1[i]; }
    return v;
}

// B-fragment: 32x16 bf16, column N = lane%16; lane<16 holds K[0..7],K[16..23],
// lane>=16 holds K[8..15],K[24..31] — two 16B loads from a K-contiguous row.
__device__ __forceinline__ v16bf load_frag_b(const __bf16* __restrict__ B,
                                             int ldb, int n0, int k0) {
    int lane = threadIdx.x & 31;
    int half = lane >> 4, c = lane & 15;
    const __bf16* p = B + (size_t)(n0 + c) * ldb + k0 + half * 8;
    bf16x8 c0 = *(const bf16x8*)(p);
    bf16x8 c1 = *(const bf16x8*)(p + 16);
    v16bf v;
#pragma unroll
    for (int i = 0; i < 8; ++i) { v[i] = c0[i]; v[i + 8] = c1[i]; }
    return v;
}

// ---------------- f32 -> bf16 convert ----------------
__global__ void cvt_bf16_kernel(const float* __restrict__ in,
                                __bf16* __restrict__ out, size_t n) {
    size_t i = (size_t)blockIdx.x * blockDim.x + threadIdx.x;
    if (i < n) out[i] = (__bf16)in[i];
}

// ---------------- WMMA GEMM: C(f32,MxN) = A(bf16,MxK) * W(bf16,NxK)^T ------
// Block = 128x256 C tile, 8 wave32s in a 2(M)x4(N) grid; each wave owns a
// 64x64 tile = 4x4 WMMA accumulators. 16 b128 loads feed 16 WMMA per K-step
// (1.0 loads/WMMA, ~32 FLOP per L2 byte) — the whole working set lives in
// MI455X's 192MB L2, so L2 (not LDS) provides the reuse.
__global__ __launch_bounds__(256) void gemm_bf16_wmma(
    const __bf16* __restrict__ A, const __bf16* __restrict__ W,
    float* __restrict__ C, int M, int N, int K) {
    int wave = threadIdx.x >> 5, lane = threadIdx.x & 31;
    int wm = wave & 1, wn = wave >> 1;
    int m0 = blockIdx.y * 128 + wm * 64;
    int n0 = blockIdx.x * 256 + wn * 64;
    v8f acc[4][4] = {};
    for (int k0 = 0; k0 < K; k0 += 32) {
        __builtin_prefetch(A + (size_t)m0 * K + k0 + 64, 0, 1);
        v16bf a[4];
#pragma unroll
        for (int i = 0; i < 4; ++i)
            a[i] = load_frag_a(A, K, m0 + i * 16, k0);
#pragma unroll
        for (int j = 0; j < 4; ++j) {
            v16bf b = load_frag_b(W, K, n0 + j * 16, k0);
#pragma unroll
            for (int i = 0; i < 4; ++i)
                acc[i][j] = wmma_bf16(a[i], b, acc[i][j]);
        }
    }
    int half = lane >> 4, c = lane & 15;
#pragma unroll
    for (int i = 0; i < 4; ++i)
#pragma unroll
        for (int j = 0; j < 4; ++j)
#pragma unroll
            for (int r = 0; r < 8; ++r)
                C[(size_t)(m0 + i * 16 + r + 8 * half) * N + n0 + j * 16 + c] =
                    acc[i][j][r];
}

// ---------------- RMS-norm + RoPE + even/odd head split ----------------
// In: (S, 32, 64) f32.  Out1/Out2: (16, S, 64) bf16 (heads 2h / 2h+1).
__global__ __launch_bounds__(32) void rms_rope_split_kernel(
    const float* __restrict__ X, const int* __restrict__ xpos,
    __bf16* __restrict__ out1, __bf16* __restrict__ out2) {
    int s = blockIdx.x, h = blockIdx.y, lane = threadIdx.x;
    const float* p = X + ((size_t)s * 32 + h) * QD;
    float x1 = p[lane], x2 = p[lane + 32];
    float ss = x1 * x1 + x2 * x2;
#pragma unroll
    for (int off = 1; off < 32; off <<= 1) ss += __shfl_xor(ss, off, 32);
    float inv = rsqrtf(ss * (1.0f / QD) + F32_EPS);
    x1 *= inv; x2 *= inv;
    float invf = __powf(10000.0f, -((float)(2 * lane)) / (float)QD);
    float f = (float)xpos[s] * invf;
    float sn, cs;
    __sincosf(f, &sn, &cs);
    float o1 =  x1 * cs + x2 * sn;
    float o2 = -x1 * sn + x2 * cs;
    __bf16* dst = ((h & 1) ? out2 : out1) + ((size_t)(h >> 1) * S_LEN + s) * QD;
    dst[lane] = (__bf16)o1;
    dst[lane + 32] = (__bf16)o2;
}

// ---------------- V transpose: (S,16,128) f32 -> (16,128,S) bf16 ----------
__global__ void build_vt_kernel(const float* __restrict__ V,
                                __bf16* __restrict__ Vt) {
    size_t idx = (size_t)blockIdx.x * blockDim.x + threadIdx.x; // (h*128+d)*S+s
    int s = (int)(idx & (S_LEN - 1));
    size_t t = idx >> 11;
    int d = (int)(t & 127);
    int h = (int)(t >> 7);
    Vt[idx] = (__bf16)V[((size_t)s * NHEADS + h) * HD + d];
}

// ---------------- lambda scalar ----------------
__global__ __launch_bounds__(32) void lam_kernel(
    const float* __restrict__ lq1, const float* __restrict__ lk1,
    const float* __restrict__ lq2, const float* __restrict__ lk2,
    float* __restrict__ lam) {
    int l = threadIdx.x;
    float d1 = lq1[l] * lk1[l] + lq1[l + 32] * lk1[l + 32];
    float d2 = lq2[l] * lk2[l] + lq2[l + 32] * lk2[l + 32];
#pragma unroll
    for (int off = 1; off < 32; off <<= 1) {
        d1 += __shfl_xor(d1, off, 32);
        d2 += __shfl_xor(d2, off, 32);
    }
    if (l == 0) *lam = __expf(d1) - __expf(d2) + LAMBDA_INIT;
}

// ---------------- Flash attention (causal, online softmax, WMMA) ----------
// Q,K: (16, S, 64) bf16.  Vt: (16, 128, S) bf16.  O: (16, S, 128) f32.
// 8 waves/block; each wave owns one 16-query tile; key chunks of 32.
__global__ __launch_bounds__(256) void flash_attn_wmma(
    const __bf16* __restrict__ Q, const __bf16* __restrict__ Kb,
    const __bf16* __restrict__ Vt, float* __restrict__ O) {
    __shared__ __bf16 plds[8][16 * 32]; // per-wave private P staging (1KB/wave)
    int wave = threadIdx.x >> 5, lane = threadIdx.x & 31;
    int half = lane >> 4, c = lane & 15;
    int h = blockIdx.y;
    int q0 = blockIdx.x * 128 + wave * 16;
    const __bf16* Qh = Q  + (size_t)h * S_LEN * QD;
    const __bf16* Kh = Kb + (size_t)h * S_LEN * QD;
    const __bf16* Vh = Vt + (size_t)h * HD * S_LEN;

    v16bf qa0 = load_frag_a(Qh, QD, q0, 0);
    v16bf qa1 = load_frag_a(Qh, QD, q0, 32);

    v8f o[8] = {};
    float mrow[8], lrow[8];
#pragma unroll
    for (int r = 0; r < 8; ++r) { mrow[r] = -1e30f; lrow[r] = 0.0f; }

    int qhi = q0 + 15;
    for (int kc = 0; kc <= qhi; kc += 32) {
        // --- scores: two 16x16 tiles over QD=64 (2 WMMA k-steps each) ---
        v8f s0 = {}, s1 = {};
        {
            v16bf b00 = load_frag_b(Kh, QD, kc, 0);
            v16bf b01 = load_frag_b(Kh, QD, kc, 32);
            s0 = wmma_bf16(qa0, b00, s0);
            s0 = wmma_bf16(qa1, b01, s0);
            v16bf b10 = load_frag_b(Kh, QD, kc + 16, 0);
            v16bf b11 = load_frag_b(Kh, QD, kc + 16, 32);
            s1 = wmma_bf16(qa0, b10, s1);
            s1 = wmma_bf16(qa1, b11, s1);
        }
        // --- scale + causal mask + per-row max candidates ---
        float pm[8];
#pragma unroll
        for (int r = 0; r < 8; ++r) {
            int qrow = q0 + r + 8 * half;
            float v0 = ((kc + c)      <= qrow) ? s0[r] * SCALE : -1e30f;
            float v1 = ((kc + 16 + c) <= qrow) ? s1[r] * SCALE : -1e30f;
            s0[r] = v0; s1[r] = v1;
            pm[r] = fmaxf(v0, v1);
        }
#pragma unroll
        for (int off = 1; off < 16; off <<= 1)
#pragma unroll
            for (int r = 0; r < 8; ++r)
                pm[r] = fmaxf(pm[r], __shfl_xor(pm[r], off, 32));
        // --- online softmax update ---
        float rs[8];
#pragma unroll
        for (int r = 0; r < 8; ++r) {
            float mnew = fmaxf(mrow[r], pm[r]);
            float scl = __expf(mrow[r] - mnew);
            float p0 = __expf(s0[r] - mnew);
            float p1 = __expf(s1[r] - mnew);
            s0[r] = p0; s1[r] = p1;
            rs[r] = p0 + p1;
            mrow[r] = mnew;
            lrow[r] *= scl;
#pragma unroll
            for (int dt = 0; dt < 8; ++dt) o[dt][r] *= scl;
        }
#pragma unroll
        for (int off = 1; off < 16; off <<= 1)
#pragma unroll
            for (int r = 0; r < 8; ++r) rs[r] += __shfl_xor(rs[r], off, 32);
#pragma unroll
        for (int r = 0; r < 8; ++r) lrow[r] += rs[r];
        // --- P (C-layout) -> LDS -> A-fragment layout ---
        __bf16* pw = plds[wave];
#pragma unroll
        for (int r = 0; r < 8; ++r) {
            int row = r + 8 * half;
            pw[row * 32 + c]      = (__bf16)s0[r];
            pw[row * 32 + 16 + c] = (__bf16)s1[r];
        }
        asm volatile("s_wait_dscnt 0" ::: "memory"); // per-wave DS counter fence
        v16bf pa;
        {
            const __bf16* pp = pw + (lane & 15) * 32 + half * 8;
            bf16x8 c0 = *(const bf16x8*)(pp);
            bf16x8 c1 = *(const bf16x8*)(pp + 16);
#pragma unroll
            for (int i = 0; i < 8; ++i) { pa[i] = c0[i]; pa[i + 8] = c1[i]; }
        }
        // --- PV: 8 d-tiles, K=32 keys, B from Vt (d-major, s contiguous) ---
#pragma unroll
        for (int dt = 0; dt < 8; ++dt) {
            v16bf vb = load_frag_b(Vh, S_LEN, dt * 16, kc);
            o[dt] = wmma_bf16(pa, vb, o[dt]);
        }
    }
    // --- epilogue: normalize and store f32 ---
#pragma unroll
    for (int dt = 0; dt < 8; ++dt)
#pragma unroll
        for (int r = 0; r < 8; ++r) {
            int qrow = q0 + r + 8 * half;
            O[((size_t)h * S_LEN + qrow) * HD + dt * 16 + c] =
                o[dt][r] / lrow[r];
        }
}

// ---------------- combine: RMS(a1 - lam*a2) * (1-LAMBDA_INIT) -> bf16 -----
// a1,a2: (16, S, 128) f32.  Out: (S, 16*128) bf16 (transposed for O-proj).
__global__ __launch_bounds__(32) void combine_kernel(
    const float* __restrict__ a1, const float* __restrict__ a2,
    const float* __restrict__ lamp, __bf16* __restrict__ out) {
    int s = blockIdx.x, h = blockIdx.y, lane = threadIdx.x;
    float lam = *lamp;
    const float* p1 = a1 + ((size_t)h * S_LEN + s) * HD;
    const float* p2 = a2 + ((size_t)h * S_LEN + s) * HD;
    float v[4], ss = 0.0f;
#pragma unroll
    for (int i = 0; i < 4; ++i) {
        int d = lane + i * 32;
        v[i] = p1[d] - lam * p2[d];
        ss += v[i] * v[i];
    }
#pragma unroll
    for (int off = 1; off < 32; off <<= 1) ss += __shfl_xor(ss, off, 32);
    float inv = rsqrtf(ss * (1.0f / HD) + F32_EPS) * (1.0f - LAMBDA_INIT);
    __bf16* dst = out + (size_t)s * HID + h * HD;
#pragma unroll
    for (int i = 0; i < 4; ++i) dst[lane + i * 32] = (__bf16)(v[i] * inv);
}

// ---------------- host-side orchestration ----------------
extern "C" void kernel_launch(void* const* d_in, const int* in_sizes, int n_in,
                              void* d_out, int out_size, void* d_ws,
                              size_t ws_size, hipStream_t stream) {
    (void)in_sizes; (void)n_in; (void)out_size; (void)ws_size;
    const float* x   = (const float*)d_in[0];
    const int*   xp  = (const int*)  d_in[1];
    const float* Wq  = (const float*)d_in[2];
    const float* Wk  = (const float*)d_in[3];
    const float* Wv  = (const float*)d_in[4];
    const float* Wo  = (const float*)d_in[5];
    const float* lq1 = (const float*)d_in[6];
    const float* lk1 = (const float*)d_in[7];
    const float* lq2 = (const float*)d_in[8];
    const float* lk2 = (const float*)d_in[9];
    float* out = (float*)d_out;

    char* ws = (char*)d_ws;
    const size_t MB = 1024ull * 1024ull;
    __bf16* XB  = (__bf16*)(ws + 0 * MB);    // 8 MB   x bf16
    __bf16* WQB = (__bf16*)(ws + 8 * MB);    // 8 MB
    __bf16* WKB = (__bf16*)(ws + 16 * MB);   // 8 MB
    __bf16* WVB = (__bf16*)(ws + 24 * MB);   // 8 MB
    __bf16* WOB = (__bf16*)(ws + 32 * MB);   // 8 MB
    float*  QF  = (float*)(ws + 40 * MB);    // 16 MB  q f32 (reused as A1)
    float*  KF  = (float*)(ws + 56 * MB);    // 16 MB  k f32 (reused as A2)
    float*  VF  = (float*)(ws + 72 * MB);    // 16 MB  v f32 (reused as AIN)
    __bf16* Q1  = (__bf16*)(ws + 88 * MB);   // 4 MB
    __bf16* Q2  = (__bf16*)(ws + 92 * MB);   // 4 MB
    __bf16* K1  = (__bf16*)(ws + 96 * MB);   // 4 MB
    __bf16* K2  = (__bf16*)(ws + 100 * MB);  // 4 MB
    __bf16* VT  = (__bf16*)(ws + 104 * MB);  // 8 MB
    float*  A1  = QF;                        // reuse after RoPE consumes QF
    float*  A2  = KF;
    __bf16* AIN = (__bf16*)VF;               // reuse after VT built
    float*  LAM = (float*)(ws + 112 * MB);

    const size_t NEL = (size_t)HID * HID;    // 2048*2048
    dim3 cvtGrid((unsigned)((NEL + 255) / 256));
    cvt_bf16_kernel<<<cvtGrid, 256, 0, stream>>>(x,  XB,  NEL);
    cvt_bf16_kernel<<<cvtGrid, 256, 0, stream>>>(Wq, WQB, NEL);
    cvt_bf16_kernel<<<cvtGrid, 256, 0, stream>>>(Wk, WKB, NEL);
    cvt_bf16_kernel<<<cvtGrid, 256, 0, stream>>>(Wv, WVB, NEL);
    cvt_bf16_kernel<<<cvtGrid, 256, 0, stream>>>(Wo, WOB, NEL);

    dim3 gGrid(HID / 256, S_LEN / 128);      // 128x256 block tiles
    gemm_bf16_wmma<<<gGrid, 256, 0, stream>>>(XB, WQB, QF, S_LEN, HID, HID);
    gemm_bf16_wmma<<<gGrid, 256, 0, stream>>>(XB, WKB, KF, S_LEN, HID, HID);
    gemm_bf16_wmma<<<gGrid, 256, 0, stream>>>(XB, WVB, VF, S_LEN, HID, HID);

    dim3 rGrid(S_LEN, 32);
    rms_rope_split_kernel<<<rGrid, 32, 0, stream>>>(QF, xp, Q1, Q2);
    rms_rope_split_kernel<<<rGrid, 32, 0, stream>>>(KF, xp, K1, K2);

    build_vt_kernel<<<(NHEADS * HD * S_LEN) / 256, 256, 0, stream>>>(VF, VT);
    lam_kernel<<<1, 32, 0, stream>>>(lq1, lk1, lq2, lk2, LAM);

    dim3 aGrid(S_LEN / 128, NHEADS);
    flash_attn_wmma<<<aGrid, 256, 0, stream>>>(Q1, K1, VT, A1);
    flash_attn_wmma<<<aGrid, 256, 0, stream>>>(Q2, K2, VT, A2);

    dim3 cGrid(S_LEN, NHEADS);
    combine_kernel<<<cGrid, 32, 0, stream>>>(A1, A2, LAM, AIN);

    gemm_bf16_wmma<<<gGrid, 256, 0, stream>>>(AIN, WOB, out, S_LEN, HID, HID);
}
